// ChunkedSlidingWindowModel_38474317038291
// MI455X (gfx1250) — matrix operand order repaired and checked by reading, not compiled, not run on hardware
//
#include <hip/hip_runtime.h>

typedef __attribute__((ext_vector_type(16))) _Float16 v16h;
typedef __attribute__((ext_vector_type(8)))  _Float16 v8h;
typedef __attribute__((ext_vector_type(8)))  float    v8f;

#define WMMA_F16(a, b, c) \
  __builtin_amdgcn_wmma_f32_16x16x32_f16(false, (a), false, (b), (short)0, (c), false, false)

// CDNA5 async global->LDS path (confirmed present on this toolchain).
#if __has_builtin(__builtin_amdgcn_global_load_async_to_lds_b128) && \
    __has_builtin(__builtin_amdgcn_s_wait_asynccnt)
#define USE_ASYNC_LDS 1
// Builtin signature (from hipcc diagnostic): arg0 = AS1 pointer to
// int __attribute__((vector_size(16))), arg1 = AS3 pointer, then imm offset, imm cpol.
typedef int v4i_vs __attribute__((vector_size(16)));
typedef __attribute__((address_space(1))) v4i_vs gv4i_t;
typedef __attribute__((address_space(3))) v4i_vs lv4i_t;
#else
#define USE_ASYNC_LDS 0
#endif

static constexpr int DM   = 512;
static constexpr int SEQ  = 8192;
static constexpr int BATCH= 2;
static constexpr int NH   = 8;
static constexpr int HD   = 64;
static constexpr int WIN  = 256;
static constexpr int NB   = SEQ / WIN;      // 32
static constexpr int MTOT = BATCH * SEQ;    // 16384
static constexpr int VTP  = 528;            // padded LDS row stride (halfs)

__device__ inline v8f vzero8() { v8f z = {}; return z; }

// ---------------------------------------------------------------------------
// Y[M,N] = A[M,K] @ B[N,K]^T ; B always fp32 weights; A fp32 or f16; Y f16/f32.
// Block = 8 waves -> 64(M) x 128(N); per K-step both tiles are staged in LDS
// via async global->LDS, then each wave runs 2x2 WMMA tiles.
// ---------------------------------------------------------------------------
template <typename AT, typename OT>
__global__ __launch_bounds__(256)
void gemm_nt(const AT* __restrict__ A, const float* __restrict__ Bw,
             OT* __restrict__ Y, int M, int N, int K) {
  constexpr int BK       = 32;
  constexpr int BLK_M    = 64;
  constexpr int BLK_N    = 128;
  constexpr int A_ROWB   = BK * (int)sizeof(AT);   // 128 (f32) / 64 (f16)
  constexpr int A_STRIDE = A_ROWB + 16;            // pad to dodge bank conflicts
  constexpr int B_ROWB   = BK * 4;                 // 128
  constexpr int B_STRIDE = B_ROWB + 16;            // 144
  constexpr int ASEG     = A_ROWB / 16;
  constexpr int BSEG     = B_ROWB / 16;
  constexpr int NCOPY    = BLK_M * ASEG + BLK_N * BSEG;

  __shared__ __align__(16) char lds[BLK_M * A_STRIDE + BLK_N * B_STRIDE];
  char* As = lds;
  char* Bs = lds + BLK_M * A_STRIDE;

  const int tid  = threadIdx.x;
  const int lane = tid & 31;
  const int wid  = tid >> 5;
  const int hi   = lane >> 4;
  const int lo   = lane & 15;
  const int mblk = blockIdx.y * BLK_M;
  const int nblk = blockIdx.x * BLK_N;
  const int wm   = (wid >> 2) * 32;   // wave M offset inside block
  const int wn   = (wid & 3)  * 32;   // wave N offset inside block

  v8f acc[2][2];
  for (int i = 0; i < 2; ++i)
    for (int j = 0; j < 2; ++j) acc[i][j] = vzero8();

  for (int kc = 0; kc < K; kc += BK) {
    // ---- stage A (64 x 32) and B (128 x 32) tiles into LDS ----
    for (int idx = tid; idx < NCOPY; idx += 256) {
      const char* gp;
      char* lp;
      int stepb;   // bytes to next K-step for prefetch
      if (idx < BLK_M * ASEG) {
        const int row = idx / ASEG, seg = idx % ASEG;
        gp = (const char*)(A + (size_t)(mblk + row) * K + kc) + seg * 16;
        lp = As + row * A_STRIDE + seg * 16;
        stepb = BK * (int)sizeof(AT);
      } else {
        const int j = idx - BLK_M * ASEG;
        const int row = j / BSEG, seg = j % BSEG;
        gp = (const char*)(Bw + (size_t)(nblk + row) * K + kc) + seg * 16;
        lp = Bs + row * B_STRIDE + seg * 16;
        stepb = BK * 4;
      }
#if USE_ASYNC_LDS
      __builtin_amdgcn_global_load_async_to_lds_b128(
          (gv4i_t*)const_cast<char*>(gp), (lv4i_t*)lp, 0, 0);
#else
      *(float4*)lp = *(const float4*)gp;
#endif
      if (kc + BK < K) __builtin_prefetch(gp + stepb, 0, 1);
    }
#if USE_ASYNC_LDS
    __builtin_amdgcn_s_wait_asynccnt(0);
#endif
    __syncthreads();

    // ---- fragments from LDS ----
    v16h af[2], bf[2];
    for (int mt = 0; mt < 2; ++mt) {
      const int row = wm + mt * 16 + lo;
      const AT* ap = (const AT*)(As + row * A_STRIDE) + hi * 16;
      #pragma unroll
      for (int i = 0; i < 8; ++i) af[mt][i]     = (_Float16)ap[i];
      #pragma unroll
      for (int i = 0; i < 8; ++i) af[mt][8 + i] = (_Float16)ap[8 + i];
    }
    for (int nt = 0; nt < 2; ++nt) {
      const int col = wn + nt * 16 + lo;
      const float* bp = (const float*)(Bs + col * B_STRIDE) + hi * 16;
      #pragma unroll
      for (int i = 0; i < 16; ++i) bf[nt][i] = (_Float16)bp[i];
    }
    for (int mt = 0; mt < 2; ++mt)
      for (int nt = 0; nt < 2; ++nt)
        acc[mt][nt] = WMMA_F16(af[mt], bf[nt], acc[mt][nt]);

    __syncthreads();   // protect LDS tiles before next staging
  }

  for (int mt = 0; mt < 2; ++mt)
    for (int nt = 0; nt < 2; ++nt) {
      const int col = nblk + wn + nt * 16 + lo;
      #pragma unroll
      for (int r = 0; r < 8; ++r) {
        const int row = mblk + wm + mt * 16 + r + hi * 8;
        Y[(size_t)row * N + col] = (OT)acc[mt][nt][r];
      }
    }
}

// ---------------------------------------------------------------------------
// Sliding-window attention: one workgroup (512 thr = 16 waves) per (b,h,block).
// Each wave owns ONE 16-row WMMA q-tile -> low VGPR pressure (no spills/MSB),
// flash-style online softmax over 32-key chunks.
// ---------------------------------------------------------------------------
__global__ __launch_bounds__(512)
void swa_kernel(const _Float16* __restrict__ q,
                const _Float16* __restrict__ k,
                const _Float16* __restrict__ v,
                _Float16* __restrict__ o) {
  extern __shared__ _Float16 smem[];
  _Float16* Vt = smem;                 // [64][VTP] transposed V tile
  _Float16* Pl = smem + 64 * VTP;      // 16 waves * 16*32 P staging

  const int bid = blockIdx.x;
  const int n = bid & (NB - 1);
  const int h = (bid >> 5) & (NH - 1);
  const int b = bid >> 8;

  const int tid  = threadIdx.x;
  const int lane = tid & 31;
  const int wid  = tid >> 5;          // 0..15
  const int hi   = lane >> 4;
  const int lo   = lane & 15;

  const size_t rowbase = (size_t)b * SEQ;
  const int base_key_s = (n - 1) * WIN;    // may be -256 for block 0
  const int qb = wid * 16;                 // this wave's local query base

  // Stage V transposed into LDS: Vt[d][key]
  for (int idx = tid; idx < 512 * 64; idx += 512) {
    const int key = idx >> 6;
    const int d   = idx & 63;
    int s = base_key_s + key; if (s < 0) s = 0;   // masked anyway
    Vt[d * VTP + key] = v[(rowbase + s) * DM + h * HD + d];
  }
  __syncthreads();

  // Persistent Q fragments: one q-tile x 2 d-chunks
  v16h qa[2];
  {
    const int qrow = n * WIN + qb + lo;
    const _Float16* qp = q + (rowbase + qrow) * DM + h * HD;
    for (int dc = 0; dc < 2; ++dc) {
      const _Float16* p0 = qp + dc * 32 + hi * 16;
      v8h a0 = *(const v8h*)(p0);
      v8h a1 = *(const v8h*)(p0 + 8);
      #pragma unroll
      for (int i = 0; i < 8; ++i) { qa[dc][i] = a0[i]; qa[dc][8 + i] = a1[i]; }
    }
  }

  v8f out[4];
  float mrow[8], lrow[8];
  for (int dt = 0; dt < 4; ++dt) out[dt] = vzero8();
  for (int r = 0; r < 8; ++r) { mrow[r] = -1e30f; lrow[r] = 0.f; }

  _Float16* pme = Pl + wid * 512;   // this wave's 16x32 P staging

  // Chunks overlapping valid keys ki in [qb+1, qb+15+WIN]: always 9 chunks.
  const int kclo = (qb + 1) >> 5;
  const int kchi = (qb + 15 + WIN) >> 5;
  for (int kc = kclo; kc <= kchi; ++kc) {
    // K fragments: 2 key tiles x 2 d-chunks (B-layout: lane n = key, K = d)
    v16h kb[2][2];
    for (int t = 0; t < 2; ++t) {
      const int keyl = kc * 32 + t * 16 + lo;
      int s = base_key_s + keyl; if (s < 0) s = 0;
      const _Float16* kp = k + (rowbase + s) * DM + h * HD;
      for (int dc = 0; dc < 2; ++dc)
        kb[t][dc] = *(const v16h*)(kp + dc * 32 + hi * 16);
    }

    // ----- scores: S = Q @ K^T (inner d = 64, two WMMA steps) -----
    v8f sc[2];
    for (int t = 0; t < 2; ++t) {
      v8f c = vzero8();
      c = WMMA_F16(qa[0], kb[t][0], c);
      c = WMMA_F16(qa[1], kb[t][1], c);
      sc[t] = c;
    }
    // ----- mask + scale + per-row chunk max -----
    float cmax[8];
    #pragma unroll
    for (int r = 0; r < 8; ++r) {
      const int qi = qb + r + hi * 8;
      float best = -1e30f;
      #pragma unroll
      for (int t = 0; t < 2; ++t) {
        const int ki = kc * 32 + t * 16 + lo;
        float sv = sc[t][r] * 0.125f;   // 1/sqrt(64)
        const bool valid = (ki >= qi + 1) && (ki <= qi + WIN) && (n > 0 || ki >= WIN);
        sv = valid ? sv : -1e30f;
        sc[t][r] = sv;
        best = fmaxf(best, sv);
      }
      cmax[r] = best;
    }
    #pragma unroll
    for (int r = 0; r < 8; ++r) {
      float x = cmax[r];
      x = fmaxf(x, __shfl_xor(x, 1, 32));
      x = fmaxf(x, __shfl_xor(x, 2, 32));
      x = fmaxf(x, __shfl_xor(x, 4, 32));
      x = fmaxf(x, __shfl_xor(x, 8, 32));
      cmax[r] = x;
    }
    // ----- online softmax update -----
    float scale8[8];
    #pragma unroll
    for (int r = 0; r < 8; ++r) {
      const float mnew = fmaxf(mrow[r], cmax[r]);
      const float sc0  = __expf(mrow[r] - mnew);
      mrow[r] = mnew;
      scale8[r] = sc0;
      float rs = 0.f;
      #pragma unroll
      for (int t = 0; t < 2; ++t) {
        const float sv = sc[t][r];
        const float p = (sv <= -1e29f) ? 0.f : __expf(sv - mnew);
        sc[t][r] = p;
        rs += p;
      }
      rs += __shfl_xor(rs, 1, 32);
      rs += __shfl_xor(rs, 2, 32);
      rs += __shfl_xor(rs, 4, 32);
      rs += __shfl_xor(rs, 8, 32);
      lrow[r] = lrow[r] * sc0 + rs;
    }
    for (int dt = 0; dt < 4; ++dt)
      #pragma unroll
      for (int r = 0; r < 8; ++r) out[dt][r] *= scale8[r];

    // ----- P: C-layout -> LDS row-major [16][32] -> A-layout fragment -----
    #pragma unroll
    for (int t = 0; t < 2; ++t)
      #pragma unroll
      for (int r = 0; r < 8; ++r)
        pme[(r + hi * 8) * 32 + t * 16 + lo] = (_Float16)sc[t][r];

    v16h pa;
    {
      const _Float16* pp = pme + lo * 32 + hi * 8;
      v8h a0 = *(const v8h*)(pp);
      v8h a1 = *(const v8h*)(pp + 16);
      #pragma unroll
      for (int i = 0; i < 8; ++i) { pa[i] = a0[i]; pa[8 + i] = a1[i]; }
    }

    // ----- out += P @ V (inner = 32 keys; B from transposed V in LDS) -----
    for (int dt = 0; dt < 4; ++dt) {
      const _Float16* vp = Vt + (dt * 16 + lo) * VTP + kc * 32 + hi * 8;
      v8h b0 = *(const v8h*)(vp);
      v8h b1 = *(const v8h*)(vp + 16);
      v16h vb;
      #pragma unroll
      for (int i = 0; i < 8; ++i) { vb[i] = b0[i]; vb[8 + i] = b1[i]; }
      out[dt] = WMMA_F16(pa, vb, out[dt]);
    }
  }

  // ----- normalize and store (f16, [s, h*64+d] layout for O projection) -----
  #pragma unroll
  for (int r = 0; r < 8; ++r) {
    const int qrow = n * WIN + qb + r + hi * 8;
    const float rl = 1.f / lrow[r];
    _Float16* op = o + (rowbase + qrow) * DM + h * HD;
    for (int dt = 0; dt < 4; ++dt)
      op[dt * 16 + lo] = (_Float16)(out[dt][r] * rl);
  }
}

// ---------------------------------------------------------------------------
extern "C" void kernel_launch(void* const* d_in, const int* in_sizes, int n_in,
                              void* d_out, int out_size, void* d_ws, size_t ws_size,
                              hipStream_t stream) {
  const float* x  = (const float*)d_in[0];
  const float* Wq = (const float*)d_in[1];
  const float* Wk = (const float*)d_in[2];
  const float* Wv = (const float*)d_in[3];
  const float* Wo = (const float*)d_in[4];
  float* out = (float*)d_out;

  _Float16* qws = (_Float16*)d_ws;
  _Float16* kws = qws + (size_t)MTOT * DM;
  _Float16* vws = kws + (size_t)MTOT * DM;
  _Float16* aws = vws + (size_t)MTOT * DM;

  const dim3 blk(256);
  const dim3 gproj(DM / 128, MTOT / 64);   // (4, 256)

  gemm_nt<float, _Float16><<<gproj, blk, 0, stream>>>(x, Wq, qws, MTOT, DM, DM);
  gemm_nt<float, _Float16><<<gproj, blk, 0, stream>>>(x, Wk, kws, MTOT, DM, DM);
  gemm_nt<float, _Float16><<<gproj, blk, 0, stream>>>(x, Wv, vws, MTOT, DM, DM);

  const size_t shbytes = (size_t)(64 * VTP + 16 * 512) * sizeof(_Float16); // ~82 KB
  swa_kernel<<<dim3(BATCH * NH * NB), dim3(512), shbytes, stream>>>(qws, kws, vws, aws);

  gemm_nt<_Float16, float><<<gproj, blk, 0, stream>>>(aws, Wo, out, MTOT, DM, DM);
}
